// Readout_40922448396322
// MI455X (gfx1250) — matrix operand order repaired and checked by reading, not compiled, not run on hardware
//
#include <hip/hip_runtime.h>

// ---------------------------------------------------------------------------
// B=128, N=64, D=128, E=32, H=256 ; P = N*N = 4096 ; F = 2D+E = 288
// Decomposition: concat@W1 + b1 = edges@W1e + Xi[i] + XjF[j], where
//   Xi  = nodes @ W1[0:128,:] + b1          (linear f32, per-node)
//   XjF = nodes @ W1[160:288,:]             (stored in WMMA C-fragment layout)
// Hot kernel layer1 is a K=32 GEMM with XjF as the WMMA C input.
// ---------------------------------------------------------------------------
#define NB   128
#define NN   64
#define DD   128
#define EE   32
#define HH   256
#define PP   4096

#define KSTEP2   8          // 256 / 32
#define NTILES   16         // 256 / 16
#define FRAG_U32 256        // one 16x32 bf16 B-fragment (8 vgprs * 32 lanes)
#define W1_ENT   (9 * NTILES * FRAG_U32)        // 36864 u32
#define W2_ENT   (KSTEP2 * NTILES * FRAG_U32)   // 32768 u32
#define XI_OFF   (W1_ENT + W2_ENT)              // u32 offset of Xi (f32)
#define XI_ENT   (NB * NN * HH)                 // 2,097,152 floats
#define XJ_OFF   (XI_OFF + XI_ENT)              // XjF, C-fragment layout
// total ws need: (XJ_OFF + XI_ENT)*4 = 17,055,744 bytes

#define LSTRIDE  260        // LDS row stride in u16 (520 B, bank-skewed)

typedef __attribute__((ext_vector_type(16))) __bf16 v16bf;
typedef __attribute__((ext_vector_type(8)))  float  v8f;
typedef __attribute__((ext_vector_type(8)))  unsigned int v8u;

// f32 pair -> packed bf16 (RNE) in one VALU op. Non-volatile: pure value
// computation, scheduler is free to reorder/batch around it.
__device__ __forceinline__ unsigned bf16pk(float x, float y) {
    unsigned r;
    asm("v_cvt_pk_bf16_f32 %0, %1, %2" : "=v"(r) : "v"(x), "v"(y));
    return r;
}
__device__ __forceinline__ unsigned short bf16one(float x) {
    unsigned r;
    asm("v_cvt_pk_bf16_f32 %0, %1, 0" : "=v"(r) : "v"(x));
    return (unsigned short)r;
}
__device__ __forceinline__ float bf16lo(unsigned w) { return __uint_as_float(w << 16); }
__device__ __forceinline__ float bf16hi(unsigned w) { return __uint_as_float(w & 0xffff0000u); }

// A-fragment K offset for bf16 16x32 A tile (ISA 7.12.2)
__device__ __forceinline__ int amapK(int v, int half) {
    return (v < 4) ? (8 * half + 2 * v) : (16 + 8 * half + 2 * (v - 4));
}

// ---------------------------------------------------------------------------
// Prep: f32 W1 [288x256] / W2 [256x256] -> bf16 B-fragments, LANE-MAJOR:
//   frag(s,t) base = (s*16+t)*256 u32 ; entry = lane*8 + v
//   lane L -> col n = t*16 + (L&15), half = L>>4 ; K = s*32 + amapK(v, half)
// ---------------------------------------------------------------------------
__global__ void prep_weights_kernel(const float* __restrict__ W1,
                                    const float* __restrict__ W2,
                                    unsigned* __restrict__ ws) {
    int idx = blockIdx.x * 256 + threadIdx.x;
    const float* W;
    int frag, e;
    unsigned* dst;
    if (idx < W1_ENT) {
        W = W1; frag = idx >> 8; e = idx & 255; dst = ws + idx;
    } else {
        int id2 = idx - W1_ENT;
        W = W2; frag = id2 >> 8; e = id2 & 255; dst = ws + W1_ENT + id2;
    }
    int s = frag >> 4, t = frag & 15;
    int L = e >> 3, v = e & 7;                 // lane-major
    int half = L >> 4;
    int n = t * 16 + (L & 15);
    int K = s * 32 + amapK(v, half);
    float lo = W[(size_t)K * HH + n];
    float hi = W[(size_t)(K + 1) * HH + n];
    *dst = bf16pk(lo, hi);
}

// ---------------------------------------------------------------------------
// Xi / XjF precompute with WMMA. One block per batch; 8 waves:
//   kind = wave&1 (0: Xi uses W1 k-steps 0..3, +b1, linear layout ;
//                  1: Xj uses k-steps 5..8, stored as C-fragments)
//   Mtile = wave>>1 -> node rows [16*Mtile, +16) ; full 256 cols (16 tiles)
// XjF layout: [b][Mtile(4)][t(16)][lane(32)][r(8)] f32
// ---------------------------------------------------------------------------
__global__ void __launch_bounds__(256, 1)
xij_kernel(const float* __restrict__ nodes,
           const float* __restrict__ b1,
           const unsigned* __restrict__ ws,
           float* __restrict__ Xi,
           float* __restrict__ XjF) {
    const int tid  = threadIdx.x;
    const int lane = tid & 31;
    const int wave = tid >> 5;
    const int kind  = wave & 1;
    const int Mtile = wave >> 1;
    const int half = lane >> 4;
    const int m16  = lane & 15;
    const int b = blockIdx.x;

    const float* nodeRow = nodes + ((size_t)b * NN + 16 * Mtile + m16) * DD;
    const unsigned* wsW1 = ws;

    v8f acc[16];
    #pragma unroll
    for (int t = 0; t < 16; ++t) acc[t] = (v8f){0,0,0,0,0,0,0,0};

    #pragma unroll
    for (int sp = 0; sp < 4; ++sp) {
        const int s = kind ? (sp + 5) : sp;     // W1 k-step index
        // phase 1: issue all A-element loads; phase 2: batch converts
        float2 f[8];
        #pragma unroll
        for (int v = 0; v < 8; ++v)
            f[v] = *(const float2*)(nodeRow + sp * 32 + amapK(v, half));
        v8u a;
        #pragma unroll
        for (int v = 0; v < 8; ++v) a[v] = bf16pk(f[v].x, f[v].y);
        v16bf af = __builtin_bit_cast(v16bf, a);

        // B fragments in groups of 4: one load clause, then 4 WMMAs
        #pragma unroll
        for (int tg = 0; tg < 4; ++tg) {
            v8u bq[4];
            #pragma unroll
            for (int q = 0; q < 4; ++q) {
                int t = tg * 4 + q;
                bq[q] = *(const v8u*)(wsW1 + (size_t)(s * NTILES + t) * FRAG_U32 + lane * 8);
            }
            #pragma unroll
            for (int q = 0; q < 4; ++q) {
                int t = tg * 4 + q;
                acc[t] = __builtin_amdgcn_wmma_f32_16x16x32_bf16(
                    false, af, false, __builtin_bit_cast(v16bf, bq[q]),
                    (short)0, acc[t], false, false);
            }
        }
    }

    if (kind == 0) {
        // Xi: linear [node][n] f32 with b1 folded in
        float* dst = Xi + (size_t)b * NN * HH;
        #pragma unroll
        for (int t = 0; t < 16; ++t) {
            int n = t * 16 + m16;
            float bias = b1[n];
            #pragma unroll
            for (int r = 0; r < 8; ++r) {
                int row = 16 * Mtile + r + 8 * half;
                dst[(size_t)row * HH + n] = acc[t][r] + bias;
            }
        }
    } else {
        // XjF: raw C-fragment dump, 32 contiguous bytes per lane per tile
        float* dst = XjF + (((size_t)b * 4 + Mtile) * NTILES) * 256;
        #pragma unroll
        for (int t = 0; t < 16; ++t)
            *(v8f*)(dst + (size_t)t * 256 + lane * 8) = acc[t];
    }
}

// ---------------------------------------------------------------------------
// Fused pair MLP. One block = 64 pairs (fixed i, j=0..63) of one batch.
// 8 waves: wave w -> rows [16*(w>>1), +16), columns [128*(w&1), +128).
// ---------------------------------------------------------------------------
__global__ void __launch_bounds__(256, 2)
mlp_pair_kernel(const float* __restrict__ edges,   // [B,4096,32]
                const float* __restrict__ b2,      // [256]
                const float* __restrict__ W3,      // [256]
                const float* __restrict__ b3,      // [1]
                const unsigned* __restrict__ ws,   // frags + Xi + XjF
                float* __restrict__ out) {         // [B,4096]
    __shared__ unsigned short sBuf[64 * LSTRIDE];  // 33,280 B

    const int tid  = threadIdx.x;
    const int lane = tid & 31;
    const int wave = tid >> 5;
    const int wrow = wave >> 1;
    const int tbase = (wave & 1) * 8;
    const int half = lane >> 4;
    const int m16  = lane & 15;

    const int bidx = blockIdx.x;
    const int b  = bidx >> 6;
    const int i  = bidx & 63;
    const int r0 = i * 64;

    const int myrow = 16 * wrow + m16;             // j for this lane's A rows
    const float* edgeR = edges + ((size_t)b * PP + r0 + myrow) * EE;

    const unsigned* wsW1 = ws;
    const unsigned* wsW2 = ws + W1_ENT;
    const float* Xi  = (const float*)(ws + XI_OFF) + ((size_t)b * NN + i) * HH;
    const float* XjF = (const float*)(ws + XJ_OFF) +
                       (((size_t)b * 4 + wrow) * NTILES) * 256;

    v8f acc[8];

    // -------- Stage 1: A1 = relu(WMMA(edges, W1e, C=XjF) + Xi[i]) --------
    {
        float2 f[8];
        #pragma unroll
        for (int v = 0; v < 8; ++v)
            f[v] = *(const float2*)(edgeR + amapK(v, half));
        v8u a;
        #pragma unroll
        for (int v = 0; v < 8; ++v) a[v] = bf16pk(f[v].x, f[v].y);
        v16bf af = __builtin_bit_cast(v16bf, a);

        #pragma unroll
        for (int tg = 0; tg < 2; ++tg) {
            v8u bq[4];
            v8f cin[4];
            #pragma unroll
            for (int q = 0; q < 4; ++q) {
                int t = tbase + tg * 4 + q;
                bq[q]  = *(const v8u*)(wsW1 + (size_t)(4 * NTILES + t) * FRAG_U32 + lane * 8);
                cin[q] = *(const v8f*)(XjF + (size_t)t * 256 + lane * 8);
            }
            #pragma unroll
            for (int q = 0; q < 4; ++q) {
                acc[tg * 4 + q] = __builtin_amdgcn_wmma_f32_16x16x32_bf16(
                    false, af, false, __builtin_bit_cast(v16bf, bq[q]),
                    (short)0, cin[q], false, false);
            }
        }
    }

    #pragma unroll
    for (int tt = 0; tt < 8; ++tt) {
        int n = (tbase + tt) * 16 + m16;
        float base = Xi[n];                        // b1 already folded in
        #pragma unroll
        for (int r = 0; r < 8; ++r) {
            int m = 16 * wrow + r + 8 * half;      // row in block == j
            float v = fmaxf(acc[tt][r] + base, 0.0f);
            sBuf[m * LSTRIDE + n] = bf16one(v);
        }
    }
    __syncthreads();

    // -------- Stage 2: A2 = relu(A1 @ W2 + b2) --------
    #pragma unroll
    for (int tt = 0; tt < 8; ++tt) acc[tt] = (v8f){0,0,0,0,0,0,0,0};

    #pragma unroll
    for (int s = 0; s < KSTEP2; ++s) {
        v8u a;
        #pragma unroll
        for (int v = 0; v < 8; ++v) {
            int K = s * 32 + amapK(v, half);
            int m = 16 * wrow + m16;
            a[v] = *(const unsigned*)&sBuf[m * LSTRIDE + K];  // ds_load_b32
        }
        v16bf af = __builtin_bit_cast(v16bf, a);

        // batch all 8 B-fragments of this k-step, then the 8-WMMA chain
        v8u bq[8];
        #pragma unroll
        for (int tt = 0; tt < 8; ++tt) {
            int t = tbase + tt;
            bq[tt] = *(const v8u*)(wsW2 + (size_t)(s * NTILES + t) * FRAG_U32 + lane * 8);
        }
        #pragma unroll
        for (int tt = 0; tt < 8; ++tt) {
            acc[tt] = __builtin_amdgcn_wmma_f32_16x16x32_bf16(
                false, af, false, __builtin_bit_cast(v16bf, bq[tt]),
                (short)0, acc[tt], false, false);
        }
    }

    __syncthreads();  // all reads of A1 done before overwrite

    #pragma unroll
    for (int tt = 0; tt < 8; ++tt) {
        int n = (tbase + tt) * 16 + m16;
        float bias = b2[n];
        #pragma unroll
        for (int r = 0; r < 8; ++r) {
            int m = 16 * wrow + r + 8 * half;
            float v = fmaxf(acc[tt][r] + bias, 0.0f);
            sBuf[m * LSTRIDE + n] = bf16one(v);
        }
    }
    __syncthreads();

    // -------- Stage 3: out = A2 @ W3 + b3 --------
    if (tid < 64) {
        int m = tid;
        float sum = b3[0];
        #pragma unroll 8
        for (int k = 0; k < HH; k += 2) {
            unsigned w = *(const unsigned*)&sBuf[m * LSTRIDE + k];
            sum += bf16lo(w) * W3[k];
            sum += bf16hi(w) * W3[k + 1];
        }
        out[(size_t)b * PP + r0 + m] = sum;
    }
}

// ---------------------------------------------------------------------------
extern "C" void kernel_launch(void* const* d_in, const int* in_sizes, int n_in,
                              void* d_out, int out_size, void* d_ws, size_t ws_size,
                              hipStream_t stream) {
    const float* nodes = (const float*)d_in[0];
    const float* edges = (const float*)d_in[1];
    const float* W1    = (const float*)d_in[2];
    const float* b1    = (const float*)d_in[3];
    const float* W2    = (const float*)d_in[4];
    const float* b2    = (const float*)d_in[5];
    const float* W3    = (const float*)d_in[6];
    const float* b3    = (const float*)d_in[7];
    unsigned* ws = (unsigned*)d_ws;     // needs ~17.1 MB
    float* Xi  = (float*)(ws + XI_OFF);
    float* XjF = (float*)(ws + XJ_OFF);

    // 1) weight swizzle to bf16 WMMA B-fragments (lane-major)
    int prep_blocks = (W1_ENT + W2_ENT) / 256;   // 272
    prep_weights_kernel<<<prep_blocks, 256, 0, stream>>>(W1, W2, ws);

    // 2) per-node partial products Xi (+b1, linear) / XjF (C-fragment layout)
    xij_kernel<<<NB, 256, 0, stream>>>(nodes, b1, ws, Xi, XjF);

    // 3) fused pair-MLP: layer1 = WMMA(edges, W1e, C=XjF) + Xi
    mlp_pair_kernel<<<NB * NN, 256, 0, stream>>>(edges, b2, W3, b3,
                                                 ws, (float*)d_out);
}
